// LiquidMedicalAI_52312701665973
// MI455X (gfx1250) — matrix operand order repaired and checked by reading, not compiled, not run on hardware
//
#include <hip/hip_runtime.h>
#include <math.h>

#define F_DIM 256
#define H_DIM 128

typedef __attribute__((ext_vector_type(16))) _Float16 v16h;
typedef __attribute__((ext_vector_type(8)))  _Float16 v8h;
typedef __attribute__((ext_vector_type(8)))  float    v8f;

#define LOG2E 1.4426950408889634f

// fast transcendentals on the TRANS pipe (v_exp_f32 / v_rcp_f32)
__device__ __forceinline__ float fast_exp(float x) {
  return __builtin_amdgcn_exp2f(x * LOG2E);
}
__device__ __forceinline__ float fast_tanh(float x) {
  // 1 - 2/(1+e^{2x}); exact limits at +/-inf, branch-free
  float e = __builtin_amdgcn_exp2f(x * (2.0f * LOG2E));
  return 1.0f - 2.0f * __builtin_amdgcn_rcpf(1.0f + e);
}
__device__ __forceinline__ float fast_sigmoid(float z) {
  float e = __builtin_amdgcn_exp2f(-z * LOG2E);
  return __builtin_amdgcn_rcpf(1.0f + e);
}

// ---- A-matrix (16x32 f16) per-lane layout (ISA 7.12.2):
// lane L (g = L>>4): halves 0..7 = K(g*8..g*8+7), halves 8..15 = K(16+g*8..16+g*8+7)
__device__ __forceinline__ v16h load_a_from_f32(const float* __restrict__ p, int g) {
  const float* c0 = p + g * 8;
  const float* c1 = p + 16 + g * 8;
  float4 f0 = ((const float4*)c0)[0];
  float4 f1 = ((const float4*)c0)[1];
  float4 f2 = ((const float4*)c1)[0];
  float4 f3 = ((const float4*)c1)[1];
  union { v16h v; _Float16 e[16]; } u;
  u.e[0]  = (_Float16)f0.x; u.e[1]  = (_Float16)f0.y; u.e[2]  = (_Float16)f0.z; u.e[3]  = (_Float16)f0.w;
  u.e[4]  = (_Float16)f1.x; u.e[5]  = (_Float16)f1.y; u.e[6]  = (_Float16)f1.z; u.e[7]  = (_Float16)f1.w;
  u.e[8]  = (_Float16)f2.x; u.e[9]  = (_Float16)f2.y; u.e[10] = (_Float16)f2.z; u.e[11] = (_Float16)f2.w;
  u.e[12] = (_Float16)f3.x; u.e[13] = (_Float16)f3.y; u.e[14] = (_Float16)f3.z; u.e[15] = (_Float16)f3.w;
  return u.v;
}

__device__ __forceinline__ v16h load_a_from_f16(const _Float16* __restrict__ p, int g) {
  v8h lo = *(const v8h*)(p + g * 8);
  v8h hi = *(const v8h*)(p + 16 + g * 8);
  return __builtin_shufflevector(lo, hi, 0,1,2,3,4,5,6,7,8,9,10,11,12,13,14,15);
}

// ---- B-matrix (32x16 f16): lane L = column n (L&15), g = L>>4 selects K half:
// halves 0..7 = K(g*8..g*8+7), halves 8..15 = K(16+g*8..16+g*8+7)
__device__ __forceinline__ v16h load_b_from_f16(const _Float16* __restrict__ p, int g) {
  v8h lo = *(const v8h*)(p + g * 8);
  v8h hi = *(const v8h*)(p + 16 + g * 8);
  return __builtin_shufflevector(lo, hi, 0,1,2,3,4,5,6,7,8,9,10,11,12,13,14,15);
}

__device__ __forceinline__ v8f wmma16(v16h a, v16h b, v8f c) {
  return __builtin_amdgcn_wmma_f32_16x16x32_f16(false, a, false, b, (short)0, c, false, false);
}

// ---- kernel 1: fp32 -> f16 weight conversion into workspace --------------
__global__ void convert_weights_kernel(const float* __restrict__ Ws,
                                       const float* __restrict__ Wih,
                                       const float* __restrict__ Whh,
                                       _Float16* __restrict__ w16) {
  const int nS = H_DIM * F_DIM;   // 32768
  const int nH = H_DIM * H_DIM;   // 16384
  int i = blockIdx.x * blockDim.x + threadIdx.x;
  if (i < nS)                w16[i] = (_Float16)Ws[i];
  else if (i < nS + nH)      w16[i] = (_Float16)Wih[i - nS];
  else if (i < nS + 2 * nH)  w16[i] = (_Float16)Whh[i - nS - nH];
}

// ---- kernel 2: fully fused pipeline, 2x row-blocked per wave -------------
__global__ __launch_bounds__(256)
void liquid_fused_kernel(const float* __restrict__ x,
                         const float* __restrict__ hs,
                         const float* __restrict__ cv,
                         const float* __restrict__ b_sensor,
                         const float* __restrict__ b_ih,
                         const float* __restrict__ b_hh,
                         const float* __restrict__ W_cls,
                         const float* __restrict__ b_cls,
                         const _Float16* __restrict__ w16,
                         float* __restrict__ out,
                         int Btot) {
  // per-wave gated-hidden slab: 32 rows x 128 cols f16; 8 waves -> 64 KB
  __shared__ _Float16 ldsH[8 * 32 * H_DIM];

  const _Float16* wS  = w16;                       // [H, F]
  const _Float16* wIH = w16 + H_DIM * F_DIM;       // [H, H]
  const _Float16* wHH = wIH + H_DIM * H_DIM;       // [H, H]

  const int tid  = threadIdx.x;
  const int wave = tid >> 5;
  const int lane = tid & 31;
  const int lm   = lane & 15;
  const int g    = lane >> 4;

  const int rowBase = blockIdx.x * 256 + wave * 32;   // 32-row slab per wave
  const int r0 = rowBase + lm;                        // A row, tile 0
  const int r1 = rowBase + 16 + lm;                   // A row, tile 1

  const float var   = cv[0];
  const float scale = (var > 2.84f) ? 0.829f : 1.0f;  // topological gate

  v8f zero = {};
  v8f acc[2][8];
  #pragma unroll
  for (int i = 0; i < 2; ++i)
    #pragma unroll
    for (int t = 0; t < 8; ++t) acc[i][t] = zero;

  // ---------------- GEMM1: raw = x @ Ws^T (one B load feeds 2 A tiles) ----
  const float* xr0 = x + (size_t)r0 * F_DIM;
  const float* xr1 = x + (size_t)r1 * F_DIM;
  for (int kk = 0; kk < F_DIM; kk += 32) {
    __builtin_prefetch(xr0 + kk + 32, 0, 1);
    __builtin_prefetch(xr1 + kk + 32, 0, 1);
    v16h a0 = load_a_from_f32(xr0 + kk, g);
    v16h a1 = load_a_from_f32(xr1 + kk, g);
    #pragma unroll
    for (int t = 0; t < 8; ++t) {
      v16h b = load_b_from_f16(wS + (size_t)(t * 16 + lm) * F_DIM + kk, g);
      acc[0][t] = wmma16(a0, b, acc[0][t]);
      acc[1][t] = wmma16(a1, b, acc[1][t]);
    }
  }

  // bias + ReLU + gate, stage to LDS f16 (D-layout -> A-layout transpose)
  _Float16* slab = ldsH + wave * (32 * H_DIM);
  #pragma unroll
  for (int t = 0; t < 8; ++t) {
    float bsv = b_sensor[t * 16 + lm];
    #pragma unroll
    for (int i = 0; i < 2; ++i) {
      #pragma unroll
      for (int v = 0; v < 8; ++v) {
        float val = acc[i][t][v] + bsv;          // element (m = i*16+v+8g, n = t*16+lm)
        val = fmaxf(val, 0.0f) * scale;
        slab[(i * 16 + v + 8 * g) * H_DIM + t * 16 + lm] = (_Float16)val;
      }
    }
  }
  __syncthreads();

  // ---------------- GEMM2: gated @ Wih^T + h @ Whh^T ----------------
  #pragma unroll
  for (int i = 0; i < 2; ++i)
    #pragma unroll
    for (int t = 0; t < 8; ++t) acc[i][t] = zero;

  const float* hr0 = hs + (size_t)r0 * H_DIM;
  const float* hr1 = hs + (size_t)r1 * H_DIM;
  for (int kk = 0; kk < H_DIM; kk += 32) {
    v16h ga0 = load_a_from_f16(slab + lm * H_DIM + kk, g);          // gated tile 0
    v16h ga1 = load_a_from_f16(slab + (16 + lm) * H_DIM + kk, g);   // gated tile 1
    v16h ha0 = load_a_from_f32(hr0 + kk, g);                        // hidden tile 0
    v16h ha1 = load_a_from_f32(hr1 + kk, g);                        // hidden tile 1
    #pragma unroll
    for (int t = 0; t < 8; ++t) {
      v16h b1 = load_b_from_f16(wIH + (size_t)(t * 16 + lm) * H_DIM + kk, g);
      acc[0][t] = wmma16(ga0, b1, acc[0][t]);
      acc[1][t] = wmma16(ga1, b1, acc[1][t]);
      v16h b2 = load_b_from_f16(wHH + (size_t)(t * 16 + lm) * H_DIM + kk, g);
      acc[0][t] = wmma16(ha0, b2, acc[0][t]);
      acc[1][t] = wmma16(ha1, b2, acc[1][t]);
    }
  }

  // ---------------- tanh + store new_hidden + classifier ----------------
  float* diag = out;              // [B]   (diagnosis)
  float* outH = out + Btot;       // [B,H] (new_hidden)

  float part[2][8];
  #pragma unroll
  for (int i = 0; i < 2; ++i)
    #pragma unroll
    for (int v = 0; v < 8; ++v) part[i][v] = 0.0f;

  #pragma unroll
  for (int t = 0; t < 8; ++t) {
    const int n    = t * 16 + lm;
    const float bb = b_ih[n] + b_hh[n];
    const float wc = W_cls[n];
    #pragma unroll
    for (int i = 0; i < 2; ++i) {
      #pragma unroll
      for (int v = 0; v < 8; ++v) {
        float hv = fast_tanh(acc[i][t][v] + bb);
        outH[(size_t)(rowBase + i * 16 + v + 8 * g) * H_DIM + n] = hv;
        part[i][v] += hv * wc;
      }
    }
  }

  // reduce over n (masks <= 8 never cross the 16-lane half of wave32)
  #pragma unroll
  for (int i = 0; i < 2; ++i) {
    #pragma unroll
    for (int v = 0; v < 8; ++v) {
      part[i][v] += __shfl_xor(part[i][v], 8);
      part[i][v] += __shfl_xor(part[i][v], 4);
      part[i][v] += __shfl_xor(part[i][v], 2);
      part[i][v] += __shfl_xor(part[i][v], 1);
    }
  }
  if (lm == 0) {
    const float bc = b_cls[0];
    #pragma unroll
    for (int i = 0; i < 2; ++i) {
      #pragma unroll
      for (int v = 0; v < 8; ++v) {
        diag[rowBase + i * 16 + v + 8 * g] = fast_sigmoid(part[i][v] + bc);
      }
    }
  }
}

extern "C" void kernel_launch(void* const* d_in, const int* in_sizes, int n_in,
                              void* d_out, int out_size, void* d_ws, size_t ws_size,
                              hipStream_t stream) {
  const float* x    = (const float*)d_in[0];
  const float* hs   = (const float*)d_in[1];
  const float* cv   = (const float*)d_in[2];
  const float* Ws   = (const float*)d_in[3];
  const float* bs   = (const float*)d_in[4];
  const float* Wih  = (const float*)d_in[5];
  const float* bih  = (const float*)d_in[6];
  const float* Whh  = (const float*)d_in[7];
  const float* bhh  = (const float*)d_in[8];
  const float* Wcls = (const float*)d_in[9];
  const float* bcls = (const float*)d_in[10];

  const int Btot = in_sizes[0] / F_DIM;          // 262144
  _Float16* w16 = (_Float16*)d_ws;               // 128 KB of scratch used

  const int nConv = H_DIM * F_DIM + 2 * H_DIM * H_DIM;   // 65536
  convert_weights_kernel<<<(nConv + 255) / 256, 256, 0, stream>>>(Ws, Wih, Whh, w16);

  liquid_fused_kernel<<<Btot / 256, 256, 0, stream>>>(
      x, hs, cv, bs, bih, bhh, Wcls, bcls, w16, (float*)d_out, Btot);
}
